// NURBSLayer_72559177499290
// MI455X (gfx1250) — matrix operand order repaired and checked by reading, not compiled, not run on hardware
//
#include <hip/hip_runtime.h>
#include <hip/hip_bf16.h>

// ---------------------------------------------------------------------------
// NURBS layer for MI455X (gfx1250, wave32, WMMA).
//   GEMMs: v_wmma_f32_16x16x32_f16, 512-thread blocks (16 waves), 4 N-tiles
//   per wave, explicit cross-iteration double buffering + sched_group_barrier
//   so WMMAs overlap the next step's global loads.
// ---------------------------------------------------------------------------

typedef __attribute__((ext_vector_type(16))) _Float16 v16h;
typedef __attribute__((ext_vector_type(8)))  _Float16 v8h;
typedef __attribute__((ext_vector_type(8)))  float    v8f;

#define B_ROWS   2048
#define IN_F     128
#define N_CP     32
#define N_DP     1024
#define HID      1023   // N_DP - 1
#define HIDP     1024   // padded
#define EPS      1e-7f

// ---------------------------------------------------------------------------
// Elementwise f32 -> f16
// ---------------------------------------------------------------------------
__global__ void k_f32_to_f16(const float* __restrict__ src,
                             _Float16* __restrict__ dst, int n) {
    int i = blockIdx.x * blockDim.x + threadIdx.x;
    if (i < n) dst[i] = (_Float16)src[i];
}

// Pad-convert weight matrix [rows, cols] f32 -> [1024, cols_p] f16 (zeros pad)
__global__ void k_pad_w16(const float* __restrict__ src,
                          _Float16* __restrict__ dst,
                          int rows, int cols, int cols_p, int total) {
    int i = blockIdx.x * blockDim.x + threadIdx.x;
    if (i >= total) return;
    int r = i / cols_p;
    int c = i - r * cols_p;
    float v = (r < rows && c < cols) ? src[(size_t)r * cols + c] : 0.0f;
    dst[i] = (_Float16)v;
}

// ---------------------------------------------------------------------------
// Fused GEMM + row-softmax.
//   C[16 x 1024] = A[16 x K] * Bw[1024 x K]^T  (logits), softmax over first
//   1023 columns per row; col 1023 -> 0.
// Block: 512 threads (16 waves). Wave w owns N in [w*64, w*64+64) (4 tiles).
// A fragment (16-bit, 16x32): lane<16: K={0..7,16..23}; lane>=16:
//   K={8..15,24..31}; M = lane%16.  -> two contiguous 16B loads.
// B fragment (16-bit, 32x16): lanes 0-15: K={0..7,16..23}, lanes 16-31:
//   K={8..15,24..31}; N = lane%16.  -> two contiguous 16B loads from W[N][K].
// ---------------------------------------------------------------------------
__global__ void gemm_softmax(const _Float16* __restrict__ A, int lda,
                             const _Float16* __restrict__ Bw, int ldb,
                             const float* __restrict__ bias, int K,
                             _Float16* __restrict__ out16,
                             float* __restrict__ out32) {
    extern __shared__ float sC[];            // [16][1024] logits, 64 KB

    const int lane = threadIdx.x & 31;
    const int wave = threadIdx.x >> 5;       // 0..15
    const int half = lane >> 4;
    const int l16  = lane & 15;
    const int rowA = blockIdx.x * 16 + l16;

    const _Float16* abase = A + (size_t)rowA * lda + half * 8;
    const _Float16* bbase[4];
#pragma unroll
    for (int t = 0; t < 4; ++t)
        bbase[t] = Bw + (size_t)(wave * 64 + t * 16 + l16) * ldb + half * 8;

    v8f acc[4];
#pragma unroll
    for (int t = 0; t < 4; ++t)
#pragma unroll
        for (int r = 0; r < 8; ++r) acc[t][r] = 0.0f;

    // ---- prologue: fragments for k = 0 ----
    v8h a0 = *(const v8h*)(abase);
    v8h a1 = *(const v8h*)(abase + 16);
    v8h bl[4], bh[4];
#pragma unroll
    for (int t = 0; t < 4; ++t) {
        bl[t] = *(const v8h*)(bbase[t]);
        bh[t] = *(const v8h*)(bbase[t] + 16);
    }

    for (int k0 = 0; k0 < K; k0 += 32) {
        // ---- prefetch next K-step (branchless; wraps to 0 on last iter) ----
        int kn = (k0 + 32 < K) ? (k0 + 32) : 0;
        v8h na0 = *(const v8h*)(abase + kn);
        v8h na1 = *(const v8h*)(abase + kn + 16);
        v8h nbl[4], nbh[4];
#pragma unroll
        for (int t = 0; t < 4; ++t) {
            nbl[t] = *(const v8h*)(bbase[t] + kn);
            nbh[t] = *(const v8h*)(bbase[t] + kn + 16);
        }

        // ---- WMMAs on the current (already resident) fragments ----
        v16h a;
#pragma unroll
        for (int e = 0; e < 8; ++e) { a[e] = a0[e]; a[8 + e] = a1[e]; }

#pragma unroll
        for (int t = 0; t < 4; ++t) {
            v16h bv;
#pragma unroll
            for (int e = 0; e < 8; ++e) { bv[e] = bl[t][e]; bv[8 + e] = bh[t][e]; }
            acc[t] = __builtin_amdgcn_wmma_f32_16x16x32_f16(
                /*neg_a=*/false, a, /*neg_b=*/false, bv,
                /*c_mod=*/(short)0, acc[t],
                /*reuse_a=*/false, /*reuse_b=*/false);
        }

#if __has_builtin(__builtin_amdgcn_sched_group_barrier)
        // Steady state: 10 VMEM reads (next step) first, then 4 WMMAs (cur).
        __builtin_amdgcn_sched_group_barrier(0x020, 10, 0);  // VMEM read
        __builtin_amdgcn_sched_group_barrier(0x008, 4, 0);   // MFMA/WMMA
#endif

        // ---- rotate buffers ----
        a0 = na0; a1 = na1;
#pragma unroll
        for (int t = 0; t < 4; ++t) { bl[t] = nbl[t]; bh[t] = nbh[t]; }
    }

    // ---- spill logits (+bias) to LDS: C layout M = r + 8*half, N = l16 ----
#pragma unroll
    for (int t = 0; t < 4; ++t) {
        int n = wave * 64 + t * 16 + l16;
        float bi = (n < HID) ? bias[n] : 0.0f;
#pragma unroll
        for (int r = 0; r < 8; ++r)
            sC[(r + 8 * half) * N_DP + n] = acc[t][r] + bi;
    }
    __syncthreads();

    // ---- row softmax: one row per wave (16 waves, 16 rows) ----
    {
        const int m = wave;
        float* rowp = sC + m * N_DP;

        float mx = -3.4e38f;
        for (int n = lane; n < HID; n += 32) mx = fmaxf(mx, rowp[n]);
#pragma unroll
        for (int off = 16; off >= 1; off >>= 1)
            mx = fmaxf(mx, __shfl_xor(mx, off, 32));

        float sum = 0.0f;
        for (int n = lane; n < HID; n += 32) {
            float e = __expf(rowp[n] - mx);
            rowp[n] = e;
            sum += e;
        }
#pragma unroll
        for (int off = 16; off >= 1; off >>= 1)
            sum += __shfl_xor(sum, off, 32);
        float inv = 1.0f / sum;

        size_t grow = (size_t)(blockIdx.x * 16 + m) * N_DP;
        for (int n = lane; n < N_DP; n += 32) {
            float e = (n < HID) ? rowp[n] * inv : 0.0f;
            out16[grow + n] = (_Float16)e;
            if (out32) out32[grow + n] = e;
        }
    }
}

// ---------------------------------------------------------------------------
// Cox-de Boor basis (degree 3, 32 control points). Knots are compile-time
// constants after full unroll: [0,0,0,0, (1/28)^3..(27/28)^3, 1,1,1,1].
// In-place ascending update is safe: new N[i] reads old N[i], N[i+1].
// ---------------------------------------------------------------------------
__device__ __forceinline__ void eval_basis(float t, float* __restrict__ Nb) {
    float kn[35];
#pragma unroll
    for (int j = 0; j < 35; ++j) {
        float v;
        if (j <= 3)       v = 0.0f;
        else if (j >= 31) v = 1.0f;
        else { float x = (float)(j - 3) * (1.0f / 28.0f); v = x * x * x; }
        kn[j] = v;
    }
    float N[35];
#pragma unroll
    for (int i = 0; i < 35; ++i) {
        int i1 = (i + 1 > 34) ? 34 : i + 1;
        N[i] = (kn[i] <= t && t < kn[i1]) ? 1.0f : 0.0f;
    }
#pragma unroll
    for (int p = 1; p <= 3; ++p) {
#pragma unroll
        for (int i = 0; i < 35 - p; ++i) {
            int ip  = (i + p > 34) ? 34 : i + p;
            int i1  = (i + 1 > 34) ? 34 : i + 1;
            int ip1 = (i + p + 1 > 34) ? 34 : i + p + 1;
            float Aa = (t - kn[i])   / (kn[ip]  - kn[i]  + EPS) * N[i];
            float Bb = (kn[ip1] - t) / (kn[ip1] - kn[i1] + EPS) * N[i1];
            N[i] = Aa + Bb;
        }
    }
#pragma unroll
    for (int k = 0; k < N_CP; ++k) Nb[k] = N[k];
}

// ---------------------------------------------------------------------------
// Per-batch-row: cumsum/clip -> ub, basis, rational weighting -> dp.
// One block (256 threads) per batch row b; each thread owns n = 4t..4t+3.
// ---------------------------------------------------------------------------
__global__ void nurbs_basis_kernel(const float* __restrict__ h3f,   // [B,1024] (col 1023 == 0)
                                   const float* __restrict__ cp,    // [B,3,32]
                                   const float* __restrict__ wts,   // [B,1,32]
                                   float* __restrict__ out_dp,      // [B,3,1024]
                                   float* __restrict__ out_ub,      // [B,1024]
                                   float* __restrict__ out_iv) {    // [B,1024]
    __shared__ float s_cw[3][N_CP];
    __shared__ float s_w[N_CP];
    __shared__ float s_sum[256];

    const int b = blockIdx.x;
    const int t = threadIdx.x;

    if (t < N_CP) {
        float wv = wts[(size_t)b * N_CP + t];
        s_w[t] = wv;
#pragma unroll
        for (int d = 0; d < 3; ++d)
            s_cw[d][t] = cp[(size_t)b * 3 * N_CP + d * N_CP + t] * wv;
    }
    __syncthreads();

    // ---- load intvls (left-padded softmax row), local prefix of 4 ----
    float iv[4], pre[4];
    float run = 0.0f;
#pragma unroll
    for (int j = 0; j < 4; ++j) {
        int n = 4 * t + j;
        iv[j] = (n == 0) ? 0.0f : h3f[(size_t)b * N_DP + n - 1];
        run += iv[j];
        pre[j] = run;
    }
    s_sum[t] = run;
    __syncthreads();

    // ---- block-wide Hillis-Steele inclusive scan over 256 partial sums ----
    for (int off = 1; off < 256; off <<= 1) {
        float add = (t >= off) ? s_sum[t - off] : 0.0f;
        __syncthreads();
        s_sum[t] += add;
        __syncthreads();
    }
    float excl = s_sum[t] - run;

    const size_t obase = (size_t)b * N_DP;
    const size_t dbase = (size_t)b * 3 * N_DP;

#pragma unroll
    for (int j = 0; j < 4; ++j) {
        int n = 4 * t + j;
        float ubv = fminf(fmaxf(excl + pre[j], 0.0f), 1.0f);

        out_iv[obase + n] = iv[j];
        out_ub[obase + n] = ubv;

        float Nb[N_CP];
        eval_basis(ubv, Nb);

        float s0 = 0.0f, s1 = 0.0f, s2 = 0.0f, sw = 0.0f;
#pragma unroll
        for (int k = 0; k < N_CP; ++k) {
            float nk = Nb[k];
            sw += nk * s_w[k];
            s0 += nk * s_cw[0][k];
            s1 += nk * s_cw[1][k];
            s2 += nk * s_cw[2][k];
        }
        float inv = 1.0f / (sw + EPS);
        out_dp[dbase + 0 * N_DP + n] = s0 * inv;
        out_dp[dbase + 1 * N_DP + n] = s1 * inv;
        out_dp[dbase + 2 * N_DP + n] = s2 * inv;
    }
}

// ---------------------------------------------------------------------------
// Launch
// ---------------------------------------------------------------------------
extern "C" void kernel_launch(void* const* d_in, const int* in_sizes, int n_in,
                              void* d_out, int out_size, void* d_ws, size_t ws_size,
                              hipStream_t stream) {
    const float* input = (const float*)d_in[0];   // [2048,128]
    const float* cp    = (const float*)d_in[1];   // [2048,3,32]
    const float* wts   = (const float*)d_in[2];   // [2048,1,32]
    const float* W1    = (const float*)d_in[3];   // [1023,128]
    const float* b1    = (const float*)d_in[4];   // [1023]
    const float* W2    = (const float*)d_in[5];   // [1023,1023]
    const float* b2    = (const float*)d_in[6];
    const float* W3    = (const float*)d_in[7];   // [1023,1023]
    const float* b3    = (const float*)d_in[8];

    char* ws = (char*)d_ws;
    size_t off = 0;
    _Float16* A1h = (_Float16*)(ws + off); off += (size_t)B_ROWS * IN_F * 2;      // 512 KB
    _Float16* W1h = (_Float16*)(ws + off); off += (size_t)HIDP * IN_F * 2;        // 256 KB
    _Float16* W2h = (_Float16*)(ws + off); off += (size_t)HIDP * HIDP * 2;        // 2 MB
    _Float16* W3h = (_Float16*)(ws + off); off += (size_t)HIDP * HIDP * 2;        // 2 MB
    _Float16* h1h = (_Float16*)(ws + off); off += (size_t)B_ROWS * N_DP * 2;      // 4 MB
    _Float16* h2h = (_Float16*)(ws + off); off += (size_t)B_ROWS * N_DP * 2;      // 4 MB
    _Float16* h3h = (_Float16*)(ws + off); off += (size_t)B_ROWS * N_DP * 2;      // 4 MB (sink)
    float*    h3f = (float*)   (ws + off); off += (size_t)B_ROWS * N_DP * 4;      // 8 MB

    float* out    = (float*)d_out;
    float* out_dp = out;                                   // [2048,3,1024]
    float* out_ub = out + (size_t)B_ROWS * 3 * N_DP;       // [2048,1,1024]
    float* out_iv = out_ub + (size_t)B_ROWS * N_DP;        // [2048,1024]

    // f32 -> f16 conversions / padded weights
    {
        int n = B_ROWS * IN_F;
        k_f32_to_f16<<<(n + 255) / 256, 256, 0, stream>>>(input, A1h, n);
    }
    {
        int tot = HIDP * IN_F;
        k_pad_w16<<<(tot + 255) / 256, 256, 0, stream>>>(W1, W1h, HID, IN_F, IN_F, tot);
    }
    {
        int tot = HIDP * HIDP;
        k_pad_w16<<<(tot + 255) / 256, 256, 0, stream>>>(W2, W2h, HID, HID, HIDP, tot);
        k_pad_w16<<<(tot + 255) / 256, 256, 0, stream>>>(W3, W3h, HID, HID, HIDP, tot);
    }

    const size_t lds_bytes = 16 * N_DP * sizeof(float);   // 64 KB logits tile
    // layer 1: K = 128
    gemm_softmax<<<B_ROWS / 16, 512, lds_bytes, stream>>>(A1h, IN_F, W1h, IN_F, b1, IN_F,
                                                          h1h, nullptr);
    // layer 2: K = 1024 (padded col is zero)
    gemm_softmax<<<B_ROWS / 16, 512, lds_bytes, stream>>>(h1h, HIDP, W2h, HIDP, b2, HIDP,
                                                          h2h, nullptr);
    // layer 3: K = 1024, also write f32 for the scan kernel
    gemm_softmax<<<B_ROWS / 16, 512, lds_bytes, stream>>>(h2h, HIDP, W3h, HIDP, b3, HIDP,
                                                          h3h, h3f);

    // cumsum + clip + basis + rational blend
    nurbs_basis_kernel<<<B_ROWS, 256, 0, stream>>>(h3f, cp, wts, out_dp, out_ub, out_iv);
}
